// GraphEncoder_84877143704124
// MI455X (gfx1250) — matrix operand order repaired and checked by reading, not compiled, not run on hardware
//
#include <hip/hip_runtime.h>

typedef __attribute__((ext_vector_type(16))) __bf16 v16bf;
typedef __attribute__((ext_vector_type(8)))  __bf16 v8bf;
typedef __attribute__((ext_vector_type(8)))  float  v8f;

#define TPB 256
#define BLK_TD 4      // column tiles per block (64 cols)
#define WAVES 8       // waves per block -> 128 rows per block
#define MAX_KSTEPS 8  // K <= 256

// ---------------------------------------------------------------- utilities
__global__ void fill_f32(float* __restrict__ p, float v, long n) {
  long i = blockIdx.x * (long)blockDim.x + threadIdx.x;
  if (i < n) p[i] = v;
}

__global__ void f32_to_bf16_k(const float* __restrict__ in, __bf16* __restrict__ out, long n) {
  long i = blockIdx.x * (long)blockDim.x + threadIdx.x;
  if (i < n) out[i] = (__bf16)in[i];
}

// Pack W[K,D] (row-major f32) into bf16 WMMA B-fragment tiles:
// Bp[dt][kt][lane][e]; matches ISA 16-bit B 32x16 layout
// (half h=lane>>4: e<8 -> K=h*8+e, e>=8 -> K=16+h*8+(e-8), column N = lane&15),
// lane-major so each lane reads 32 contiguous bytes.
__global__ void pack_weight_k(const float* __restrict__ W, __bf16* __restrict__ Bp, int K, int D) {
  long i = blockIdx.x * (long)blockDim.x + threadIdx.x;
  long total = (long)K * D;
  if (i >= total) return;
  int e  = (int)(i & 15);
  int l  = (int)((i >> 4) & 31);
  long t = i >> 9;                 // tile index
  int kt = (int)(t % (K >> 5));
  int dt = (int)(t / (K >> 5));
  int k = kt * 32 + (e < 8 ? (l >> 4) * 8 + e : 16 + (l >> 4) * 8 + (e - 8));
  int d = dt * 16 + (l & 15);
  Bp[i] = (__bf16)W[(long)k * D + d];
}

// ---------------------------------------------------------------- WMMA GEMM
// C[M,D] = A[M,K](bf16,row-major) * W(packed bf16) + bias, f32 accumulate.
// Block = 8 waves; block computes 128 rows x 64 cols. B tiles for the block's
// 4 column tiles are staged once into LDS with async global->LDS DMA; each
// wave then reuses one A fragment across 4 WMMAs per K-step.
__global__ __launch_bounds__(TPB) void gemm_bias_wmma(
    const __bf16* __restrict__ A, const __bf16* __restrict__ Bp,
    const float* __restrict__ bias, float* __restrict__ C,
    int M, int K, int D) {
  __shared__ __bf16 Bs[BLK_TD * MAX_KSTEPS * 512];   // 32 KB max

  const int ksteps  = K >> 5;
  const int tilesD  = D >> 4;
  const int blocksD = tilesD / BLK_TD;
  const int bm  = blockIdx.x / blocksD;
  const int bd  = blockIdx.x % blocksD;
  const int td0 = bd * BLK_TD;

  // ---- cooperative async stage of B tiles [td0, td0+BLK_TD) into LDS ----
  const __bf16* bsrc = Bp + (long)td0 * ksteps * 512;
  const int chunks = BLK_TD * ksteps * 64;           // 16-byte chunks
  for (int c = threadIdx.x; c < chunks; c += blockDim.x) {
    unsigned      lds_off = (unsigned)(uintptr_t)(&Bs[c * 8]);
    unsigned long gaddr   = (unsigned long)(uintptr_t)(bsrc + c * 8);
    asm volatile("global_load_async_to_lds_b128 %0, %1, off"
                 :: "v"(lds_off), "v"(gaddr) : "memory");
  }
  asm volatile("s_wait_asynccnt 0" ::: "memory");
  __syncthreads();

  const int lane = threadIdx.x & 31;
  const int wv   = threadIdx.x >> 5;
  const long tm  = (long)bm * WAVES + wv;
  if (tm * 16 >= M) return;                          // after barrier: safe

  const int hi  = lane >> 4;
  int row = (int)(tm * 16) + (lane & 15);
  if (row >= M) row = M - 1;                         // M is 16-aligned here

  v8f acc0 = {0.f,0.f,0.f,0.f,0.f,0.f,0.f,0.f};
  v8f acc1 = acc0, acc2 = acc0, acc3 = acc0;

  // ISA 16-bit A 16x32 layout: half 'hi' holds K = hi*8+0..7 and 16+hi*8+0..7
  const __bf16* arow = A + (long)row * K + hi * 8;
  const __bf16* bl   = &Bs[lane << 4];               // lane's 32B within a tile

  for (int kt = 0; kt < ksteps; ++kt) {
    v8bf alo = *(const v8bf*)(arow);
    v8bf ahi = *(const v8bf*)(arow + 16);
    v16bf a;
#pragma unroll
    for (int i = 0; i < 8; ++i) { a[i] = alo[i]; a[i + 8] = ahi[i]; }
    v16bf b0 = *(const v16bf*)(bl + (0 * ksteps + kt) * 512);
    v16bf b1 = *(const v16bf*)(bl + (1 * ksteps + kt) * 512);
    v16bf b2 = *(const v16bf*)(bl + (2 * ksteps + kt) * 512);
    v16bf b3 = *(const v16bf*)(bl + (3 * ksteps + kt) * 512);
    acc0 = __builtin_amdgcn_wmma_f32_16x16x32_bf16(false, a, false, b0, (short)0, acc0, false, false);
    acc1 = __builtin_amdgcn_wmma_f32_16x16x32_bf16(false, a, false, b1, (short)0, acc1, false, false);
    acc2 = __builtin_amdgcn_wmma_f32_16x16x32_bf16(false, a, false, b2, (short)0, acc2, false, false);
    acc3 = __builtin_amdgcn_wmma_f32_16x16x32_bf16(false, a, false, b3, (short)0, acc3, false, false);
    arow += 32;
  }

  // C/D 16x16 f32 layout: N = lane&15, VGPR v -> M = hi*8 + v
  v8f accs[4] = {acc0, acc1, acc2, acc3};
#pragma unroll
  for (int j = 0; j < BLK_TD; ++j) {
    int col = (td0 + j) * 16 + (lane & 15);
    float bv = bias[col];
#pragma unroll
    for (int v = 0; v < 8; ++v) {
      int m = (int)(tm * 16) + hi * 8 + v;
      if (m < M) C[(long)m * D + col] = accs[j][v] + bv;
    }
  }
}

// ---------------------------------------------------------------- edge phase
__device__ __forceinline__ void edge_pair(const int* __restrict__ ei, int E, int e,
                                          int& s, int& d) {
  if (e < E) { s = ei[e]; d = ei[E + e]; } else { s = d = e - E; } // self-loops
}

// float atomic max via ordered-int trick (global_atomic_max_i32 / min_u32)
__device__ __forceinline__ void atomicMaxF(float* addr, float val) {
  if (val >= 0.f) atomicMax((int*)addr, __float_as_int(val));
  else            atomicMin((unsigned int*)addr, __float_as_uint(val));
}

// pass 1: one wave per (edge, head); lanes cover channels (coalesced gathers),
// wave-level shfl_xor reduction. e = att[h].leakyrelu(xl[src]+xr[dst]); max->mx
__global__ void edge_scores_wave(const float* __restrict__ xl, const float* __restrict__ xr,
                                 const int* __restrict__ ei, const float* __restrict__ att,
                                 float* __restrict__ esc, float* __restrict__ mx,
                                 int E, int EP, int H, int C) {
  long w = (blockIdx.x * (long)blockDim.x + threadIdx.x) >> 5;
  int lane = threadIdx.x & 31;
  if (w >= (long)EP * H) return;
  int h = (int)(w % H);
  int e = (int)(w / H);
  int s, d; edge_pair(ei, E, e, s, d);
  const float* pl = xl + (long)s * H * C + h * C;
  const float* pr = xr + (long)d * H * C + h * C;
  const float* pa = att + h * C;
  __builtin_prefetch(pl, 0, 1);
  __builtin_prefetch(pr, 0, 1);
  float acc = 0.f;
  for (int c = lane; c < C; c += 32) {
    float v = pl[c] + pr[c];
    v = v > 0.f ? v : 0.2f * v;
    acc += v * pa[c];
  }
#pragma unroll
  for (int o = 16; o > 0; o >>= 1) acc += __shfl_xor(acc, o, 32);
  if (lane == 0) {
    esc[w] = acc;
    atomicMaxF(&mx[(long)d * H + h], acc);
  }
}

// pass 2: ex = exp(e - mx[dst]); den[dst] += ex
__global__ void edge_exp(float* __restrict__ esc, const float* __restrict__ mx,
                         const int* __restrict__ ei, float* __restrict__ den,
                         int E, int EP, int H) {
  long t = blockIdx.x * (long)blockDim.x + threadIdx.x;
  if (t >= (long)EP * H) return;
  int h = (int)(t % H);
  int e = (int)(t / H);
  int s, d; edge_pair(ei, E, e, s, d);
  (void)s;
  float ex = __expf(esc[t] - mx[(long)d * H + h]);
  esc[t] = ex;
  atomicAdd(&den[(long)d * H + h], ex);
}

// pass 3: one thread per (edge, channel) -> coalesced atomic scatter-add
__global__ void edge_aggregate_chan(const float* __restrict__ xl, const float* __restrict__ esc,
                                    const float* __restrict__ den, const int* __restrict__ ei,
                                    float* __restrict__ out, int E, int EP, int H, int C) {
  long t = blockIdx.x * (long)blockDim.x + threadIdx.x;
  long DHC = (long)H * C;
  if (t >= (long)EP * DHC) return;
  int r = (int)(t % DHC);
  int e = (int)(t / DHC);
  int h = r / C;
  int s, d; edge_pair(ei, E, e, s, d);
  float alpha = esc[(long)e * H + h] / (den[(long)d * H + h] + 1e-16f);
  atomicAdd(&out[(long)d * DHC + r], alpha * xl[(long)s * DHC + r]);
}

// ---------------------------------------------------------------- epilogues
__global__ void elu_bias_bf16(const float* __restrict__ in, const float* __restrict__ bias,
                              __bf16* __restrict__ outb, long n, int D) {
  long i = blockIdx.x * (long)blockDim.x + threadIdx.x;
  if (i >= n) return;
  float v = in[i] + bias[i % D];
  v = v > 0.f ? v : (__expf(v) - 1.f);
  outb[i] = (__bf16)v;
}

__global__ void elu_bias_pool(const float* __restrict__ in, const float* __restrict__ bias,
                              const int* __restrict__ batch, float* __restrict__ pool,
                              float* __restrict__ cnt, long n, int D) {
  long i = blockIdx.x * (long)blockDim.x + threadIdx.x;
  if (i >= n) return;
  int c = (int)(i % D);
  long nd = i / D;
  float v = in[i] + bias[c];
  v = v > 0.f ? v : (__expf(v) - 1.f);
  int g = batch[nd];
  atomicAdd(&pool[(long)g * D + c], v);
  if (c == 0) atomicAdd(&cnt[g], 1.f);
}

__global__ void pool_norm_bf16(const float* __restrict__ pool, const float* __restrict__ cnt,
                               __bf16* __restrict__ outb, long n, int D) {
  long i = blockIdx.x * (long)blockDim.x + threadIdx.x;
  if (i >= n) return;
  outb[i] = (__bf16)(pool[i] / fmaxf(cnt[i / D], 1.f));
}

// ---------------------------------------------------------------- launch
static inline long cdiv(long a, long b) { return (a + b - 1) / b; }

extern "C" void kernel_launch(void* const* d_in, const int* in_sizes, int n_in,
                              void* d_out, int out_size, void* d_ws, size_t ws_size,
                              hipStream_t stream) {
  const int IN = 128, D1 = 256, D2 = 128, H1 = 4, C1 = 64, OUTC = 128, G = 64;

  const float* x     = (const float*)d_in[0];
  const int*   ei    = (const int*)d_in[1];
  const int*   batch = (const int*)d_in[2];
  const float* W1l   = (const float*)d_in[3];
  const float* b1l   = (const float*)d_in[4];
  const float* W1r   = (const float*)d_in[5];
  const float* b1r   = (const float*)d_in[6];
  const float* att1  = (const float*)d_in[7];
  const float* bias1 = (const float*)d_in[8];
  const float* W2l   = (const float*)d_in[9];
  const float* b2l   = (const float*)d_in[10];
  const float* W2r   = (const float*)d_in[11];
  const float* b2r   = (const float*)d_in[12];
  const float* att2  = (const float*)d_in[13];
  const float* bias2 = (const float*)d_in[14];
  const float* Wlin  = (const float*)d_in[15];
  const float* blin  = (const float*)d_in[16];

  int N  = in_sizes[0] / IN;
  int E  = in_sizes[1] / 2;
  int EP = E + N;          // edges incl. self-loops

  // workspace bump allocator (256B aligned)
  char* ws = (char*)d_ws;
  size_t off = 0;
  auto alloc = [&](size_t bytes) -> void* {
    void* p = ws + off;
    off += (bytes + 255) & ~(size_t)255;
    return p;
  };

  __bf16* xb    = (__bf16*)alloc((size_t)N * IN * 2);
  float*  xl1   = (float*) alloc((size_t)N * D1 * 4);
  float*  xr1   = (float*) alloc((size_t)N * D1 * 4);
  float*  out1  = (float*) alloc((size_t)N * D1 * 4);
  __bf16* hb    = (__bf16*)alloc((size_t)N * D1 * 2);
  float*  xl2   = (float*) alloc((size_t)N * D2 * 4);
  float*  xr2   = (float*) alloc((size_t)N * D2 * 4);
  float*  out2  = (float*) alloc((size_t)N * D2 * 4);
  float*  e1    = (float*) alloc((size_t)EP * H1 * 4);
  float*  m1    = (float*) alloc((size_t)N * H1 * 4);
  float*  den1  = (float*) alloc((size_t)N * H1 * 4);
  float*  e2    = (float*) alloc((size_t)EP * 4);
  float*  m2    = (float*) alloc((size_t)N * 4);
  float*  den2  = (float*) alloc((size_t)N * 4);
  float*  pool  = (float*) alloc((size_t)G * D2 * 4);
  float*  cnt   = (float*) alloc((size_t)G * 4);
  __bf16* poolb = (__bf16*)alloc((size_t)G * D2 * 2);
  __bf16* W1lp  = (__bf16*)alloc((size_t)IN * D1 * 2);
  __bf16* W1rp  = (__bf16*)alloc((size_t)IN * D1 * 2);
  __bf16* W2lp  = (__bf16*)alloc((size_t)D1 * D2 * 2);
  __bf16* W2rp  = (__bf16*)alloc((size_t)D1 * D2 * 2);
  __bf16* Wlp   = (__bf16*)alloc((size_t)D2 * OUTC * 2);
  (void)ws_size; (void)n_in; (void)out_size;

  // ---- weight packing (once per launch; tiny) ----
  pack_weight_k<<<cdiv((long)IN * D1, TPB), TPB, 0, stream>>>(W1l, W1lp, IN, D1);
  pack_weight_k<<<cdiv((long)IN * D1, TPB), TPB, 0, stream>>>(W1r, W1rp, IN, D1);
  pack_weight_k<<<cdiv((long)D1 * D2, TPB), TPB, 0, stream>>>(W2l, W2lp, D1, D2);
  pack_weight_k<<<cdiv((long)D1 * D2, TPB), TPB, 0, stream>>>(W2r, W2rp, D1, D2);
  pack_weight_k<<<cdiv((long)D2 * OUTC, TPB), TPB, 0, stream>>>(Wlin, Wlp, D2, OUTC);

  // ---- layer 1 ----
  f32_to_bf16_k<<<cdiv((long)N * IN, TPB), TPB, 0, stream>>>(x, xb, (long)N * IN);

  long gb1 = cdiv(N, 128) * ((D1 / 16) / BLK_TD);
  gemm_bias_wmma<<<gb1, TPB, 0, stream>>>(xb, W1lp, b1l, xl1, N, IN, D1);
  gemm_bias_wmma<<<gb1, TPB, 0, stream>>>(xb, W1rp, b1r, xr1, N, IN, D1);

  fill_f32<<<cdiv((long)N * H1, TPB), TPB, 0, stream>>>(m1, -INFINITY, (long)N * H1);
  fill_f32<<<cdiv((long)N * H1, TPB), TPB, 0, stream>>>(den1, 0.f, (long)N * H1);
  fill_f32<<<cdiv((long)N * D1, TPB), TPB, 0, stream>>>(out1, 0.f, (long)N * D1);

  long eh1 = (long)EP * H1;
  edge_scores_wave<<<cdiv(eh1 * 32, TPB), TPB, 0, stream>>>(xl1, xr1, ei, att1, e1, m1,
                                                            E, EP, H1, C1);
  edge_exp<<<cdiv(eh1, TPB), TPB, 0, stream>>>(e1, m1, ei, den1, E, EP, H1);
  edge_aggregate_chan<<<cdiv((long)EP * D1, TPB), TPB, 0, stream>>>(xl1, e1, den1, ei, out1,
                                                                    E, EP, H1, C1);

  elu_bias_bf16<<<cdiv((long)N * D1, TPB), TPB, 0, stream>>>(out1, bias1, hb, (long)N * D1, D1);

  // ---- layer 2 ----
  long gb2 = cdiv(N, 128) * ((D2 / 16) / BLK_TD);
  gemm_bias_wmma<<<gb2, TPB, 0, stream>>>(hb, W2lp, b2l, xl2, N, D1, D2);
  gemm_bias_wmma<<<gb2, TPB, 0, stream>>>(hb, W2rp, b2r, xr2, N, D1, D2);

  fill_f32<<<cdiv((long)N, TPB), TPB, 0, stream>>>(m2, -INFINITY, (long)N);
  fill_f32<<<cdiv((long)N, TPB), TPB, 0, stream>>>(den2, 0.f, (long)N);
  fill_f32<<<cdiv((long)N * D2, TPB), TPB, 0, stream>>>(out2, 0.f, (long)N * D2);

  edge_scores_wave<<<cdiv((long)EP * 32, TPB), TPB, 0, stream>>>(xl2, xr2, ei, att2, e2, m2,
                                                                 E, EP, 1, D2);
  edge_exp<<<cdiv((long)EP, TPB), TPB, 0, stream>>>(e2, m2, ei, den2, E, EP, 1);
  edge_aggregate_chan<<<cdiv((long)EP * D2, TPB), TPB, 0, stream>>>(xl2, e2, den2, ei, out2,
                                                                    E, EP, 1, D2);

  // ---- pooling + head ----
  fill_f32<<<cdiv((long)G * D2, TPB), TPB, 0, stream>>>(pool, 0.f, (long)G * D2);
  fill_f32<<<cdiv((long)G, TPB), TPB, 0, stream>>>(cnt, 0.f, (long)G);

  elu_bias_pool<<<cdiv((long)N * D2, TPB), TPB, 0, stream>>>(out2, bias2, batch, pool, cnt,
                                                             (long)N * D2, D2);
  pool_norm_bf16<<<cdiv((long)G * D2, TPB), TPB, 0, stream>>>(pool, cnt, poolb, (long)G * D2, D2);

  long gb3 = cdiv(G, 128) * ((OUTC / 16) / BLK_TD);
  gemm_bias_wmma<<<gb3, TPB, 0, stream>>>(poolb, Wlp, blin, (float*)d_out, G, D2, OUTC);
}
